// SiglipNaViTWrapper_76081050681764
// MI455X (gfx1250) — matrix operand order repaired and checked by reading, not compiled, not run on hardware
//
#include <hip/hip_runtime.h>

// ---------------------------------------------------------------------------
// SigLIP-NaViT encoder forward for gfx1250 (CDNA5, wave32, WMMA bf16).
// D=768, H=12, HD=64, FF=3072, L=4, N=2048, P=4096, NIMG=8.
// ---------------------------------------------------------------------------

#define N_TOK 2048
#define DIM   768
#define NHEAD 12
#define HDIM  64
#define FFD   3072
#define NLAY  4

typedef __attribute__((ext_vector_type(16))) __bf16           v16bf;
typedef __attribute__((ext_vector_type(8)))  float            v8f;
typedef __attribute__((ext_vector_type(16))) unsigned short   us16;
typedef __attribute__((ext_vector_type(8)))  unsigned short   us8;

// Native fp32 -> bf16 (RNE); lowers to v_cvt_pk_bf16_f32 on gfx1250.
static __device__ __forceinline__ unsigned short f2bf(float f) {
  __bf16 h = (__bf16)f;
  return __builtin_bit_cast(unsigned short, h);
}

static __device__ __forceinline__ v16bf bc16(us16 u) {
  return __builtin_bit_cast(v16bf, u);
}
static __device__ __forceinline__ v16bf join8(us8 lo, us8 hi) {
  us16 u = __builtin_shufflevector(lo, hi, 0,1,2,3,4,5,6,7,8,9,10,11,12,13,14,15);
  return __builtin_bit_cast(v16bf, u);
}
static __device__ __forceinline__ v8f wmma_bf16(v16bf a, v16bf b, v8f c) {
  return __builtin_amdgcn_wmma_f32_16x16x32_bf16(false, a, false, b,
                                                 (short)0, c, false, false);
}
static __device__ __forceinline__ float gelu_tanh(float x) {
  float x3 = x * x * x;
  return 0.5f * x * (1.0f + tanhf(0.7978845608028654f * (x + 0.044715f * x3)));
}

// ---------------------------------------------------------------------------
// seg[i] = searchsorted(cu, i, 'right') - 1   (cu has NIMG+1 = 9 entries)
// ---------------------------------------------------------------------------
__global__ void seg_k(const int* __restrict__ cu, int* __restrict__ seg, int n) {
  int i = blockIdx.x * blockDim.x + threadIdx.x;
  if (i >= n) return;
  int s = 0;
#pragma unroll
  for (int j = 1; j <= 8; ++j) s += (cu[j] <= i) ? 1 : 0;
  seg[i] = s;
}

// ---------------------------------------------------------------------------
// x[i,:] += pos_table[ids[i],:]
// ---------------------------------------------------------------------------
__global__ void posadd_k(float* __restrict__ x, const float* __restrict__ tab,
                         const int* __restrict__ ids) {
  int i = blockIdx.x;
  int p = ids[i];
#pragma unroll
  for (int j = 0; j < 3; ++j) {
    int c = threadIdx.x + j * 256;
    x[i * DIM + c] += tab[p * DIM + c];
  }
}

// ---------------------------------------------------------------------------
// LayerNorm: one wave per row (768 = 32 lanes * 24)
// ---------------------------------------------------------------------------
__global__ __launch_bounds__(256) void layernorm_k(
    const float* __restrict__ X, const float* __restrict__ g,
    const float* __restrict__ b, float* __restrict__ T) {
  int lane = threadIdx.x & 31, w = threadIdx.x >> 5;
  int row = blockIdx.x * 8 + w;
  const float* x = X + row * DIM;
  float vals[24];
  float s = 0.f;
#pragma unroll
  for (int j = 0; j < 24; ++j) { vals[j] = x[lane + 32 * j]; s += vals[j]; }
#pragma unroll
  for (int d = 1; d < 32; d <<= 1) s += __shfl_xor(s, d, 32);
  float mean = s * (1.0f / DIM);
  float vs = 0.f;
#pragma unroll
  for (int j = 0; j < 24; ++j) { float dd = vals[j] - mean; vs += dd * dd; }
#pragma unroll
  for (int d = 1; d < 32; d <<= 1) vs += __shfl_xor(vs, d, 32);
  float inv = rsqrtf(vs * (1.0f / DIM) + 1e-6f);
  float* t = T + row * DIM;
#pragma unroll
  for (int j = 0; j < 24; ++j) {
    int c = lane + 32 * j;
    t[c] = (vals[j] - mean) * inv * g[c] + b[c];
  }
}

// ---------------------------------------------------------------------------
// GEMM: C[M,N] = epi(A[M,K] @ W[N,K]^T + bias[N] (+resid))
// EPI: 0 = bias, 1 = bias + residual, 2 = gelu(bias)
// Block tile 128x64, BK=32, 256 threads (8 waves), bf16 WMMA, fp32 acc.
// ---------------------------------------------------------------------------
template <int EPI>
__global__ __launch_bounds__(256) void gemm_bias(
    const float* __restrict__ A, const float* __restrict__ W,
    const float* __restrict__ bias, const float* __restrict__ resid,
    float* __restrict__ C, int M, int N, int K) {
  __shared__ __align__(16) unsigned short sA[128][40];
  __shared__ __align__(16) unsigned short sB[64][40];

  int tid = threadIdx.x;
  int lane = tid & 31, w = tid >> 5;
  int rowTile = blockIdx.y * 128;
  int colTile = blockIdx.x * 64;
  int m = lane & 15, hb = lane >> 4;

  v8f acc[4] = {};

  int rr = tid >> 1, cc = (tid & 1) * 16;
  const float* aRow = A + (size_t)(rowTile + rr) * K + cc;
  const float* wRow = W + (size_t)(colTile + (rr & 63)) * K + cc;

  for (int k0 = 0; k0 < K; k0 += 32) {
    __syncthreads();
    {   // stage A tile 128x32 (fp32 -> bf16), 4 x b128 loads
      const float4* src = (const float4*)(aRow + k0);
#pragma unroll
      for (int i = 0; i < 4; ++i) {
        float4 f = src[i];
        sA[rr][cc + 4 * i + 0] = f2bf(f.x);
        sA[rr][cc + 4 * i + 1] = f2bf(f.y);
        sA[rr][cc + 4 * i + 2] = f2bf(f.z);
        sA[rr][cc + 4 * i + 3] = f2bf(f.w);
      }
    }
    if (tid < 128) {   // stage B tile 64x32
      const float4* src = (const float4*)(wRow + k0);
#pragma unroll
      for (int i = 0; i < 4; ++i) {
        float4 f = src[i];
        sB[rr][cc + 4 * i + 0] = f2bf(f.x);
        sB[rr][cc + 4 * i + 1] = f2bf(f.y);
        sB[rr][cc + 4 * i + 2] = f2bf(f.z);
        sB[rr][cc + 4 * i + 3] = f2bf(f.w);
      }
    }
    // prefetch next K-tile while this one is consumed
    if (k0 + 32 < K) {
      __builtin_prefetch(aRow + k0 + 32, 0, 1);
      if (tid < 128) __builtin_prefetch(wRow + k0 + 32, 0, 1);
    }
    __syncthreads();

    // A fragment: lane<16 -> row m, K {0..7,16..23}; lane>=16 -> K {8..15,24..31}
    us8 a_lo = *(const us8*)&sA[w * 16 + m][hb * 8];
    us8 a_hi = *(const us8*)&sA[w * 16 + m][hb * 8 + 16];
    v16bf afrag = join8(a_lo, a_hi);

#pragma unroll
    for (int nt = 0; nt < 4; ++nt) {
      // B fragment: lane = output column, K {0..7,16..23} / {8..15,24..31} per half-wave
      us8 b_lo = *(const us8*)&sB[nt * 16 + m][hb * 8];
      us8 b_hi = *(const us8*)&sB[nt * 16 + m][hb * 8 + 16];
      acc[nt] = wmma_bf16(afrag, join8(b_lo, b_hi), acc[nt]);
    }
  }

#pragma unroll
  for (int nt = 0; nt < 4; ++nt) {
    int col = colTile + nt * 16 + m;
    float bv = bias[col];
#pragma unroll
    for (int r = 0; r < 8; ++r) {
      int row = rowTile + w * 16 + hb * 8 + r;
      float val = acc[nt][r] + bv;
      if (EPI == 2) val = gelu_tanh(val);
      if (EPI == 1) val += resid[(size_t)row * N + col];
      C[(size_t)row * N + col] = val;
    }
  }
}

// ---------------------------------------------------------------------------
// Flash attention with block-diagonal (segment) mask.
// grid = (NHEAD, N_TOK/128), block = 256 (8 waves).
// Each wave: one 16-row query tile of one head; streams 32-key tiles.
// ---------------------------------------------------------------------------
__global__ __launch_bounds__(256) void attn_k(
    const float* __restrict__ Q, const float* __restrict__ Km,
    const float* __restrict__ V, const int* __restrict__ seg,
    float* __restrict__ O) {
  __shared__ __align__(16) unsigned short sP[8][16][40];

  int h = blockIdx.x;
  int lane = threadIdx.x & 31, w = threadIdx.x >> 5;
  int qbase = blockIdx.y * 128 + w * 16;
  int m = lane & 15, hb = lane >> 4;
  const float scale = 0.125f;  // 64^-0.5

  // Q fragments (scale folded in): two K-halves of the 16x64 head tile
  const float* qp = Q + (size_t)(qbase + m) * DIM + h * HDIM;
  us16 uq0, uq1;
#pragma unroll
  for (int i = 0; i < 8; ++i) {
    uq0[i]     = f2bf(qp[hb * 16 + i] * scale);
    uq0[8 + i] = f2bf(qp[hb * 16 + 8 + i] * scale);
    uq1[i]     = f2bf(qp[32 + hb * 16 + i] * scale);
    uq1[8 + i] = f2bf(qp[32 + hb * 16 + 8 + i] * scale);
  }
  v16bf aq0 = bc16(uq0), aq1 = bc16(uq1);

  int segq[8];
#pragma unroll
  for (int r = 0; r < 8; ++r) segq[r] = seg[qbase + hb * 8 + r];
  int segql = seg[qbase], segqh = seg[qbase + 15];

  float rm[8], rs[8];
#pragma unroll
  for (int r = 0; r < 8; ++r) { rm[r] = -1e30f; rs[r] = 0.f; }
  v8f oacc[4] = {};

  for (int kb = 0; kb < N_TOK; kb += 32) {
    // seg is monotonic: skip key tiles fully outside this query tile's range
    if (seg[kb] > segqh || seg[kb + 31] < segql) continue;

    // scores: Q(16x64) @ K^T(64x32), two 16-col tiles
    v8f sc[2];
#pragma unroll
    for (int ct = 0; ct < 2; ++ct) {
      int key = kb + ct * 16 + m;
      const float4* kp = (const float4*)(Km + (size_t)key * DIM + h * HDIM + hb * 16);
      us16 ub0, ub1;
#pragma unroll
      for (int i = 0; i < 4; ++i) {
        float4 f0 = kp[i];       // K dims hb*16 .. hb*16+15
        float4 f1 = kp[i + 8];   // K dims 32+hb*16 .. (+8 float4 = +32 floats)
        ub0[4 * i + 0] = f2bf(f0.x); ub0[4 * i + 1] = f2bf(f0.y);
        ub0[4 * i + 2] = f2bf(f0.z); ub0[4 * i + 3] = f2bf(f0.w);
        ub1[4 * i + 0] = f2bf(f1.x); ub1[4 * i + 1] = f2bf(f1.y);
        ub1[4 * i + 2] = f2bf(f1.z); ub1[4 * i + 3] = f2bf(f1.w);
      }
      v8f s = wmma_bf16(aq0, bc16(ub0), v8f{});
      sc[ct] = wmma_bf16(aq1, bc16(ub1), s);
    }

    int segk0 = seg[kb + m], segk1 = seg[kb + 16 + m];

    // online softmax (row-wise over 32 keys, rows split 8-per-halfwave)
#pragma unroll
    for (int r = 0; r < 8; ++r) {
      float s0 = (segk0 == segq[r]) ? sc[0][r] : -1e30f;
      float s1 = (segk1 == segq[r]) ? sc[1][r] : -1e30f;
      float mx = fmaxf(s0, s1);
#pragma unroll
      for (int d = 1; d < 16; d <<= 1) mx = fmaxf(mx, __shfl_xor(mx, d, 16));
      float nm = fmaxf(rm[r], mx);
      float fr = __expf(rm[r] - nm);
      float p0 = __expf(s0 - nm);
      float p1 = __expf(s1 - nm);
      float ps = p0 + p1;
#pragma unroll
      for (int d = 1; d < 16; d <<= 1) ps += __shfl_xor(ps, d, 16);
      rs[r] = rs[r] * fr + ps;
      rm[r] = nm;
      sc[0][r] = p0; sc[1][r] = p1;
#pragma unroll
      for (int nt = 0; nt < 4; ++nt) oacc[nt][r] *= fr;
    }

    // transpose P (C layout -> A layout) through per-wave LDS slab
#pragma unroll
    for (int ct = 0; ct < 2; ++ct)
#pragma unroll
      for (int r = 0; r < 8; ++r)
        sP[w][hb * 8 + r][ct * 16 + m] = f2bf(sc[ct][r]);
    asm volatile("s_wait_dscnt 0x0" ::: "memory");

    us8 p_lo = *(const us8*)&sP[w][m][hb * 16];
    us8 p_hi = *(const us8*)&sP[w][m][hb * 16 + 8];
    v16bf pfrag = join8(p_lo, p_hi);

    // out += P(16x32) @ V(32x64)
#pragma unroll
    for (int nt = 0; nt < 4; ++nt) {
      const float* vp = V + h * HDIM + nt * 16 + m;
      us16 ubv;
#pragma unroll
      for (int i = 0; i < 16; ++i)
        ubv[i] = f2bf(vp[(size_t)(kb + hb * 16 + i) * DIM]);
      oacc[nt] = wmma_bf16(pfrag, bc16(ubv), oacc[nt]);
    }
  }

#pragma unroll
  for (int nt = 0; nt < 4; ++nt)
#pragma unroll
    for (int r = 0; r < 8; ++r) {
      int row = qbase + hb * 8 + r;
      O[(size_t)row * DIM + h * HDIM + nt * 16 + m] = oacc[nt][r] / rs[r];
    }
}

// ---------------------------------------------------------------------------
// Driver
// ---------------------------------------------------------------------------
extern "C" void kernel_launch(void* const* d_in, const int* in_sizes, int n_in,
                              void* d_out, int out_size, void* d_ws, size_t ws_size,
                              hipStream_t stream) {
  const float* pv      = (const float*)d_in[0];
  const int*   pos_ids = (const int*)d_in[1];
  const int*   cu      = (const int*)d_in[2];
  // d_in[3] = max_seqlen (unused)
  const float* patch_w = (const float*)d_in[4];
  const float* patch_b = (const float*)d_in[5];
  const float* pos_tab = (const float*)d_in[6];
  const float* ln1_g   = (const float*)d_in[7];
  const float* ln1_b   = (const float*)d_in[8];
  const float* q_w     = (const float*)d_in[9];
  const float* q_b     = (const float*)d_in[10];
  const float* k_w     = (const float*)d_in[11];
  const float* k_b     = (const float*)d_in[12];
  const float* v_w     = (const float*)d_in[13];
  const float* v_b     = (const float*)d_in[14];
  const float* o_w     = (const float*)d_in[15];
  const float* o_b     = (const float*)d_in[16];
  const float* ln2_g   = (const float*)d_in[17];
  const float* ln2_b   = (const float*)d_in[18];
  const float* fc1_w   = (const float*)d_in[19];
  const float* fc1_b   = (const float*)d_in[20];
  const float* fc2_w   = (const float*)d_in[21];
  const float* fc2_b   = (const float*)d_in[22];

  const size_t ND = (size_t)N_TOK * DIM;
  float* x = (float*)d_ws;
  float* t = x + ND;
  float* q = t + ND;
  float* k = q + ND;
  float* v = k + ND;
  float* o = v + ND;
  float* g = o + ND;                       // N_TOK * FFD
  int* seg = (int*)(g + (size_t)N_TOK * FFD);

  dim3 blk(256);
  dim3 gD(DIM / 64, N_TOK / 128);          // N=768 GEMMs
  dim3 gF(FFD / 64, N_TOK / 128);          // N=3072 GEMM
  dim3 gA(NHEAD, N_TOK / 128);             // attention
  dim3 gLN(N_TOK / 8);

  seg_k<<<dim3((N_TOK + 255) / 256), blk, 0, stream>>>(cu, seg, N_TOK);

  // patch embed + positional
  gemm_bias<0><<<gD, blk, 0, stream>>>(pv, patch_w, patch_b, nullptr, x,
                                       N_TOK, DIM, DIM);
  posadd_k<<<dim3(N_TOK), blk, 0, stream>>>(x, pos_tab, pos_ids);

  for (int l = 0; l < NLAY; ++l) {
    const size_t wDD = (size_t)l * DIM * DIM;
    const size_t wFD = (size_t)l * FFD * DIM;
    layernorm_k<<<gLN, blk, 0, stream>>>(x, ln1_g + l * DIM, ln1_b + l * DIM, t);
    gemm_bias<0><<<gD, blk, 0, stream>>>(t, q_w + wDD, q_b + l * DIM, nullptr, q,
                                         N_TOK, DIM, DIM);
    gemm_bias<0><<<gD, blk, 0, stream>>>(t, k_w + wDD, k_b + l * DIM, nullptr, k,
                                         N_TOK, DIM, DIM);
    gemm_bias<0><<<gD, blk, 0, stream>>>(t, v_w + wDD, v_b + l * DIM, nullptr, v,
                                         N_TOK, DIM, DIM);
    attn_k<<<gA, blk, 0, stream>>>(q, k, v, seg, o);
    gemm_bias<1><<<gD, blk, 0, stream>>>(o, o_w + wDD, o_b + l * DIM, x, x,
                                         N_TOK, DIM, DIM);   // x += o @ ow^T + ob
    layernorm_k<<<gLN, blk, 0, stream>>>(x, ln2_g + l * DIM, ln2_b + l * DIM, t);
    gemm_bias<2><<<gF, blk, 0, stream>>>(t, fc1_w + wFD, fc1_b + (size_t)l * FFD,
                                         nullptr, g, N_TOK, FFD, DIM);
    float* dst = (l == NLAY - 1) ? (float*)d_out : x;
    gemm_bias<1><<<gD, blk, 0, stream>>>(g, fc2_w + wFD, fc2_b + l * DIM, x, dst,
                                         N_TOK, DIM, FFD);   // dst = x + g @ f2w^T + b
  }
}
